// FlashCoreAttention_45878840655884
// MI455X (gfx1250) — matrix operand order repaired and checked by reading, not compiled, not run on hardware
//
#include <hip/hip_runtime.h>
#include <math.h>

// ---------------------------------------------------------------------------
// Fused RoPE + flash attention for MI455X (gfx1250, wave32, WMMA).
//
// - QK^T computed transposed: ST = K_tile(16k x 32d) * Q^T(32d x 16q), so the
//   score tile's D-layout (lane = q-row, VGPR = key) IS the A-layout of the
//   P operand for P*V: no LDS relayout; softmax stats are per-lane scalars.
// - Software pipeline with double-buffered LDS: next chunk's raw K/V global
//   loads are issued before the current chunk's 16 WMMAs; the loadcnt wait
//   lands after compute, hiding HBM/L2 latency. One barrier per chunk.
// - Softmax in exp2 domain (log2e folded into the Q scale): every exponential
//   is a bare v_exp_f32. O-rescale skipped via wave-uniform __all() when no
//   row max changed (steady state after the first few chunks).
// - RoPE: Q fully in registers (partner dims live in the other k-step operand
//   of the same lane); K staged with in-thread dim pairing and incremental
//   angle rotation (no per-chunk transcendentals).
// ---------------------------------------------------------------------------

typedef _Float16 half_t;
typedef __attribute__((ext_vector_type(16))) _Float16 v16h;
typedef __attribute__((ext_vector_type(8)))  _Float16 v8h;
typedef __attribute__((ext_vector_type(4)))  _Float16 v4h;
typedef __attribute__((ext_vector_type(8)))  float    v8f;
typedef __attribute__((ext_vector_type(4)))  float    v4f;

#define SEQ     2048
#define BSZ     4
#define NHEAD   16
#define HDIM    64
#define CHUNK   32                    // keys per iteration
#define NCHUNK  (SEQ / CHUNK)
#define LOG_ROPE_BASE 9.210340371976184f     // ln(10000)
// 1/sqrt(64) * log2(e): scores come out of WMMA already in exp2 domain
#define QK_SCALE_LOG2 0.18033688011112042f

#define WMMA_F16(A, B, C) \
    __builtin_amdgcn_wmma_f32_16x16x32_f16(false, (A), false, (B), (short)0, (C), false, false)

// One lane's A operand: halfs [0..7] and [16..23] relative to p.
__device__ __forceinline__ v16h load_a16(const half_t* p) {
    v8h a = *(const v8h*)(p);
    v8h b = *(const v8h*)(p + 16);
    v16h r;
#pragma unroll
    for (int i = 0; i < 8; ++i) { r[i] = a[i]; r[8 + i] = b[i]; }
    return r;
}

// Broadcast per-row scalar (held at lane==row, rows 0..15) into the O/C
// per-VGPR layout: row j for lanes 0..15, row 8+j for lanes 16..31.
__device__ __forceinline__ float bcast_row(float x, int j, int hi) {
    int b0 = __builtin_amdgcn_readlane(__builtin_bit_cast(int, x), j);
    int b1 = __builtin_amdgcn_readlane(__builtin_bit_cast(int, x), 8 + j);
    return __builtin_bit_cast(float, hi ? b1 : b0);
}

__global__ __launch_bounds__(256)
void flash_rope_attn_kernel(const float* __restrict__ q,
                            const float* __restrict__ k,
                            const float* __restrict__ v,
                            float* __restrict__ out) {
    __shared__ __align__(32) half_t kh[2][32][64];    // ping-pong RoPE'd K
    __shared__ __align__(32) half_t vhT[2][64][32];   // ping-pong V^T [dim][key]

    const int tid  = threadIdx.x;
    const int lane = tid & 31;
    const int w    = tid >> 5;          // wave 0..7
    const int lo   = lane & 15;
    const int hi   = lane >> 4;

    const int bid   = blockIdx.x;
    const int tTile = bid & 7;          // 8 tiles of 256 Q rows
    const int head  = (bid >> 3) & 15;
    const int batch = bid >> 7;
    const int t0    = tTile * 256 + w * 32;   // wave owns 32 Q rows (2 m-tiles)

    const size_t rowStride = (size_t)BSZ * NHEAD * HDIM;
    const size_t headOff   = (size_t)batch * NHEAD * HDIM + (size_t)head * HDIM;

    // ---- Q -> B-layout registers, RoPE applied in-register -----------------
    v16h qb0[2], qb1[2];
#pragma unroll
    for (int mt = 0; mt < 2; ++mt) {
        const int t = t0 + mt * 16 + lo;      // q row (lane = B column = q row)
        const float* qp = q + (size_t)t * rowStride + headOff + hi * 16;
        float x[16], y[16];
#pragma unroll
        for (int i = 0; i < 16; i += 4) {
            v4f xa = *(const v4f*)(qp + i);
            v4f ya = *(const v4f*)(qp + 32 + i);
            x[i]=xa[0]; x[i+1]=xa[1]; x[i+2]=xa[2]; x[i+3]=xa[3];
            y[i]=ya[0]; y[i+1]=ya[1]; y[i+2]=ya[2]; y[i+3]=ya[3];
        }
#pragma unroll
        for (int i = 0; i < 16; ++i) {
            float invf = __expf(-(float)(hi * 16 + i) * (LOG_ROPE_BASE / 32.0f));
            float sv, cv;
            __sincosf((float)t * invf, &sv, &cv);
            qb0[mt][i] = (half_t)((x[i] * cv - y[i] * sv) * QK_SCALE_LOG2);
            qb1[mt][i] = (half_t)((y[i] * cv + x[i] * sv) * QK_SCALE_LOG2);
        }
    }

    // ---- K staging trig state: incremental rotation ------------------------
    const int krow = w * 4 + (lane >> 3);     // key row staged by this thread
    const int cw   = lane & 7;                // paired dims {cw*4+i, +32}
    float kc[4], ks[4], kdc[4], kds[4];
#pragma unroll
    for (int i = 0; i < 4; ++i) {
        float invf = __expf(-(float)(cw * 4 + i) * (LOG_ROPE_BASE / 32.0f));
        __sincosf((float)krow * invf, &ks[i], &kc[i]);   // angle at chunk 0
        __sincosf(32.0f * invf, &kds[i], &kdc[i]);       // per-chunk delta
    }

    v8f o[2][4];
#pragma unroll
    for (int a = 0; a < 2; ++a)
#pragma unroll
        for (int b = 0; b < 4; ++b) o[a][b] = (v8f){0,0,0,0,0,0,0,0};
    float mrow[2] = {-1e30f, -1e30f};
    float lrow[2] = {0.0f, 0.0f};

    // raw next-chunk K/V in flight (registers)
    v4f kx, ky, va, vb4;
    auto issue_loads = [&](int cc) {
        const float* kp = k + (size_t)(cc * CHUNK + krow) * rowStride + headOff + cw * 4;
        kx = *(const v4f*)(kp);
        ky = *(const v4f*)(kp + 32);
        const float* vp = v + (size_t)(cc * CHUNK + krow) * rowStride + headOff + cw * 8;
        va  = *(const v4f*)(vp);
        vb4 = *(const v4f*)(vp + 4);
    };
    auto stage_store = [&](int buf) {
        v4h a, b;
#pragma unroll
        for (int i = 0; i < 4; ++i) {
            a[i] = (half_t)(kx[i] * kc[i] - ky[i] * ks[i]);
            b[i] = (half_t)(ky[i] * kc[i] + kx[i] * ks[i]);
        }
        *(v4h*)&kh[buf][krow][cw * 4]      = a;
        *(v4h*)&kh[buf][krow][cw * 4 + 32] = b;
#pragma unroll
        for (int i = 0; i < 4; ++i) {        // rotate (c,s) by +32*invf
            float nc = kc[i] * kdc[i] - ks[i] * kds[i];
            ks[i]    = ks[i] * kdc[i] + kc[i] * kds[i];
            kc[i]    = nc;
        }
#pragma unroll
        for (int i = 0; i < 4; ++i) {
            vhT[buf][cw * 8 + i][krow]     = (half_t)va[i];
            vhT[buf][cw * 8 + 4 + i][krow] = (half_t)vb4[i];
        }
    };

    // prologue: fill buffer 0
    issue_loads(0);
    stage_store(0);

    for (int c = 0; c < NCHUNK; ++c) {
        __syncthreads();   // stage(c) visible; compute(c-1) done before stage(c+1)

        // issue next chunk's raw loads now; wait lands after compute
        const int cn = (c + 1 < NCHUNK) ? c + 1 : c;
        issue_loads(cn);
        if (c + 2 < NCHUNK) {                // near-scope prefetch for chunk c+2
            __builtin_prefetch(k + (size_t)((c + 2) * CHUNK + krow) * rowStride + headOff + cw * 4, 0, 3);
            __builtin_prefetch(v + (size_t)((c + 2) * CHUNK + krow) * rowStride + headOff + cw * 8, 0, 3);
        }

        const half_t (*khc)[64] = kh[c & 1];
        const half_t (*vhc)[32] = vhT[c & 1];

#pragma unroll
        for (int mt = 0; mt < 2; ++mt) {
            // ---- ST = K_tile x Q^T (scores transposed; 4 WMMAs) -----------
            v8f st0 = (v8f){0,0,0,0,0,0,0,0};
            v8f st1 = (v8f){0,0,0,0,0,0,0,0};
            {
                v16h ka;
                ka = *(const v16h*)&khc[     lo][     hi * 16]; st0 = WMMA_F16(ka, qb0[mt], st0);
                ka = *(const v16h*)&khc[     lo][32 + hi * 16]; st0 = WMMA_F16(ka, qb1[mt], st0);
                ka = *(const v16h*)&khc[16 + lo][     hi * 16]; st1 = WMMA_F16(ka, qb0[mt], st1);
                ka = *(const v16h*)&khc[16 + lo][32 + hi * 16]; st1 = WMMA_F16(ka, qb1[mt], st1);
            }
            // Lane owns q-row `lo`, keys {hi*8+j} (st0) and {16+hi*8+j} (st1):
            // exactly the A-layout slots of P. Stats are per-lane scalars.
            float tmax = fmaxf(st0[0], st1[0]);
#pragma unroll
            for (int j = 1; j < 8; ++j) tmax = fmaxf(tmax, fmaxf(st0[j], st1[j]));
            tmax = fmaxf(tmax, __shfl_xor(tmax, 16, 32));   // combine lane halves
            const float nm = fmaxf(mrow[mt], tmax);

            v16h pa;
            float rsum = 0.0f;
#pragma unroll
            for (int j = 0; j < 8; ++j) {
                float p0 = __builtin_amdgcn_exp2f(st0[j] - nm);   // bare v_exp_f32
                float p1 = __builtin_amdgcn_exp2f(st1[j] - nm);
                rsum += p0 + p1;
                pa[j]     = (half_t)p0;
                pa[8 + j] = (half_t)p1;
            }
            rsum += __shfl_xor(rsum, 16, 32);

            // ---- O rescale only when some row's max moved (rarely) --------
            if (__all(nm == mrow[mt])) {
                lrow[mt] += rsum;
            } else {
                float cf = __builtin_amdgcn_exp2f(mrow[mt] - nm);
                lrow[mt] = lrow[mt] * cf + rsum;
                mrow[mt] = nm;
#pragma unroll
                for (int j = 0; j < 8; ++j) {
                    float cj = bcast_row(cf, j, hi);
                    o[mt][0][j] *= cj; o[mt][1][j] *= cj;
                    o[mt][2][j] *= cj; o[mt][3][j] *= cj;
                }
            }
            // ---- O += P x V (4 WMMAs over 64 output dims) -----------------
#pragma unroll
            for (int dt = 0; dt < 4; ++dt) {
                v16h vb = load_a16(&vhc[dt * 16 + lo][hi * 8]);
                o[mt][dt] = WMMA_F16(pa, vb, o[mt][dt]);
            }
        }

        // convert + store next chunk into the other buffer (waits on loads here)
        stage_store((c + 1) & 1);
    }

    // ---- finalize: O /= l, write fp32 (T, B, H*D) --------------------------
#pragma unroll
    for (int mt = 0; mt < 2; ++mt) {
        float linv = 1.0f / lrow[mt];
#pragma unroll
        for (int j = 0; j < 8; ++j) {
            float lj = bcast_row(linv, j, hi);
            const int t = t0 + mt * 16 + j + 8 * hi;
            float* op = out + (size_t)t * rowStride + headOff;
            op[ 0 + lo] = o[mt][0][j] * lj;
            op[16 + lo] = o[mt][1][j] * lj;
            op[32 + lo] = o[mt][2][j] * lj;
            op[48 + lo] = o[mt][3][j] * lj;
        }
    }
}

extern "C" void kernel_launch(void* const* d_in, const int* in_sizes, int n_in,
                              void* d_out, int out_size, void* d_ws, size_t ws_size,
                              hipStream_t stream) {
    const float* q = (const float*)d_in[0];
    const float* k = (const float*)d_in[1];
    const float* v = (const float*)d_in[2];
    // masks (d_in[3], d_in[4]) are all-false and unused by the reference path
    float* out = (float*)d_out;

    dim3 grid(BSZ * NHEAD * (SEQ / 256));   // 512 workgroups
    dim3 block(256);                        // 8 waves (wave32)
    flash_rope_attn_kernel<<<grid, block, 0, stream>>>(q, k, v, out);
}
